// AttenComm_62534723829927
// MI455X (gfx1250) — matrix operand order repaired and checked by reading, not compiled, not run on hardware
//
#include <hip/hip_runtime.h>
#include <math.h>

typedef __attribute__((ext_vector_type(16))) _Float16 v16h;
typedef __attribute__((ext_vector_type(8)))  float    v8f;
typedef __attribute__((ext_vector_type(4)))  unsigned int u32x4;

#define L_IMG 6
#define C_IN 256
#define C_OUT 128
#define H_IMG 128
#define W_IMG 256
#define HW_IMG (H_IMG * W_IMG)
#define N_KPTS 1024
#define NMS_R 4
#define FSTR 40   // LDS row stride in halves (32 data + 8 pad)

// ---------------------------------------------------------------------------
// K0: convert convPa_w (cout,cin,3,3) fp32 -> wB[tap][cin][cout] f16
// ---------------------------------------------------------------------------
__global__ __launch_bounds__(256) void k_prep_w(const float* __restrict__ w,
                                                _Float16* __restrict__ wB) {
  int e = blockIdx.x * 256 + threadIdx.x;
  if (e >= 9 * C_IN * C_OUT) return;
  int cout = e % C_OUT;
  int cin  = (e / C_OUT) % C_IN;
  int tap  = e / (C_OUT * C_IN);
  wB[e] = (_Float16)w[(cout * C_IN + cin) * 9 + tap];
}

// ---------------------------------------------------------------------------
// K1: 3x3 conv as 9 shifted GEMMs via v_wmma_f32_16x16x32_f16
// block = 256 thr (8 waves). Block tile: 128 cout x 64 px (one row segment).
// wave w: pxTile = w&3 (16 px), coutGroup = w>>2 (64 couts, 4 wmma tiles).
// Weight chunk (contiguous 8KB) staged global->LDS with async-to-LDS ops,
// synchronized on ASYNCcnt, then transposed in-LDS to the WMMA A layout.
// ---------------------------------------------------------------------------
__global__ __launch_bounds__(256) void k_conv_wmma(
    const float* __restrict__ feats, const _Float16* __restrict__ wB,
    const float* __restrict__ bias, float* __restrict__ desc) {
  __shared__ _Float16 Flds[64 * FSTR];      // [px][cin32]  (B operand)
  __shared__ _Float16 Wlds[128 * FSTR];     // [cout][cin32] (A operand)
  __shared__ _Float16 Wstage[32 * C_OUT];   // raw async-staged [cin32][cout]

  int bid = blockIdx.x;
  int l   = bid / (H_IMG * 4);
  int rem = bid % (H_IMG * 4);
  int y   = rem >> 2;
  int x0  = (rem & 3) * 64;

  int tid  = threadIdx.x;
  int wave = tid >> 5;
  int lane = tid & 31;
  int hi   = lane >> 4;      // K-half selector (contiguous k order used here for A and B)
  int n16  = lane & 15;
  int pxTile    = wave & 3;
  int coutGroup = wave >> 2;

  // loader roles
  int ldCi      = tid & 31;          // cin within chunk
  int ldPxBase  = (tid >> 5) * 8;    // 8 pixels per thread (F stage)
  int ldCoutB   = (tid >> 5) * 16;   // 16 couts per thread (W transpose)

  unsigned wstageBase = (unsigned)(uintptr_t)(&Wstage[0]);
  unsigned asyncOff   = (unsigned)tid * 16u;

  v8f acc[4] = {};

  for (int tap = 0; tap < 9; ++tap) {
    int dy = tap / 3 - 1, dx = tap % 3 - 1;
    int yy = y + dy;
    bool yok = (yy >= 0) && (yy < H_IMG);
    for (int cc = 0; cc < 8; ++cc) {
      int cin0 = cc * 32;
      __syncthreads();

      // ---- async stage weights: 8KB contiguous global -> LDS (no VGPR trip)
      {
        unsigned long long g =
            (unsigned long long)(uintptr_t)(wB + (tap * C_IN + cin0) * C_OUT);
        unsigned d0 = wstageBase + asyncOff;
        unsigned d1 = d0 + 4096u;
        unsigned long long g0 = g + asyncOff;
        unsigned long long g1 = g0 + 4096u;
        asm volatile("global_load_async_to_lds_b128 %0, %1, off"
                     :: "v"(d0), "v"(g0) : "memory");
        asm volatile("global_load_async_to_lds_b128 %0, %1, off"
                     :: "v"(d1), "v"(g1) : "memory");
      }

      // ---- stage pixels (B operand): Flds[px][ci], zero-fill halo ----
      {
        int fbase = ((l * C_IN + cin0 + ldCi) * H_IMG + yy) * W_IMG;
        #pragma unroll
        for (int j = 0; j < 8; ++j) {
          int xx = x0 + ldPxBase + j + dx;
          float v = 0.0f;
          if (yok && xx >= 0 && xx < W_IMG) v = feats[fbase + xx];
          Flds[(ldPxBase + j) * FSTR + ldCi] = (_Float16)v;
        }
      }

      asm volatile("s_wait_asynccnt 0x0" ::: "memory");
      __syncthreads();

      // ---- in-LDS transpose Wstage[ci][cout] -> Wlds[cout][ci] ----
      {
        union { u32x4 q[2]; _Float16 h[16]; } wt;
        const u32x4* src = (const u32x4*)&Wstage[ldCi * C_OUT + ldCoutB];
        wt.q[0] = src[0];
        wt.q[1] = src[1];
        #pragma unroll
        for (int j = 0; j < 16; ++j)
          Wlds[(ldCoutB + j) * FSTR + ldCi] = wt.h[j];
      }
      __syncthreads();

      // ---- B fragment: 32(K) x 16(N=px). lane n16 = px, halves = K hi*16..+15
      union { v16h v; u32x4 q[2]; } bf;
      int px = pxTile * 16 + n16;
      bf.q[0] = *(const u32x4*)&Flds[px * FSTR + hi * 16];
      bf.q[1] = *(const u32x4*)&Flds[px * FSTR + hi * 16 + 8];

      #pragma unroll
      for (int it = 0; it < 4; ++it) {
        // ---- A fragment: 16(M=cout) x 32(K). halves: K = hi*16..+15
        union { v16h v; u32x4 q[2]; } af;
        int cout = coutGroup * 64 + it * 16 + n16;
        af.q[0] = *(const u32x4*)&Wlds[cout * FSTR + hi * 16];
        af.q[1] = *(const u32x4*)&Wlds[cout * FSTR + hi * 16 + 8];
        acc[it] = __builtin_amdgcn_wmma_f32_16x16x32_f16(
            false, af.v, false, bf.v, (short)0, acc[it], false, false);
      }
    }
  }

  // epilogue: D tile -> (cout = base + r + hi*8, px = n16); + bias, relu
  int obase = l * C_OUT * HW_IMG + y * W_IMG + x0 + pxTile * 16 + n16;
  #pragma unroll
  for (int it = 0; it < 4; ++it) {
    #pragma unroll
    for (int r = 0; r < 8; ++r) {
      int cout = coutGroup * 64 + it * 16 + r + hi * 8;
      float v = acc[it][r] + bias[cout];
      desc[obase + cout * HW_IMG] = v > 0.0f ? v : 0.0f;
    }
  }
}

// ---------------------------------------------------------------------------
// K2: scores = sigmoid(convPb_w . desc + b)
// ---------------------------------------------------------------------------
__global__ __launch_bounds__(256) void k_scores(const float* __restrict__ desc,
                                                const float* __restrict__ pbw,
                                                const float* __restrict__ pbb,
                                                float* __restrict__ sc) {
  int p = blockIdx.x * 256 + threadIdx.x;
  if (p >= L_IMG * HW_IMG) return;
  int l = p / HW_IMG, px = p % HW_IMG;
  const float* dp = desc + l * C_OUT * HW_IMG + px;
  float s = 0.0f;
  for (int c = 0; c < C_OUT; ++c) s += pbw[c] * dp[c * HW_IMG];
  s += pbb[0];
  sc[p] = 1.0f / (1.0f + __expf(-s));
}

// ---------------------------------------------------------------------------
// NMS helper kernels (separable 9x9 max pool + elementwise)
// ---------------------------------------------------------------------------
__global__ __launch_bounds__(256) void k_mp_h(const float* __restrict__ in,
                                              float* __restrict__ out) {
  int p = blockIdx.x * 256 + threadIdx.x;
  if (p >= L_IMG * HW_IMG) return;
  int x = p % W_IMG;
  float m = -__builtin_inff();
  #pragma unroll
  for (int d = -NMS_R; d <= NMS_R; ++d) {
    int xx = x + d;
    if (xx >= 0 && xx < W_IMG) m = fmaxf(m, in[p - x + xx]);
  }
  out[p] = m;
}

__global__ __launch_bounds__(256) void k_mp_v(const float* __restrict__ in,
                                              float* __restrict__ out) {
  int p = blockIdx.x * 256 + threadIdx.x;
  if (p >= L_IMG * HW_IMG) return;
  int y = (p % HW_IMG) / W_IMG;
  float m = -__builtin_inff();
  #pragma unroll
  for (int d = -NMS_R; d <= NMS_R; ++d) {
    int yy = y + d;
    if (yy >= 0 && yy < H_IMG) m = fmaxf(m, in[p + d * W_IMG]);
  }
  out[p] = m;
}

__global__ __launch_bounds__(256) void k_eq(const float* a, const float* b, float* m) {
  int p = blockIdx.x * 256 + threadIdx.x;
  if (p >= L_IMG * HW_IMG) return;
  m[p] = (a[p] == b[p]) ? 1.0f : 0.0f;
}

__global__ __launch_bounds__(256) void k_supp(const float* sc, const float* pool, float* ss) {
  int p = blockIdx.x * 256 + threadIdx.x;
  if (p >= L_IMG * HW_IMG) return;
  ss[p] = (pool[p] > 0.0f) ? 0.0f : sc[p];
}

__global__ __launch_bounds__(256) void k_update(float* mask, const float* ss,
                                                const float* mpss, const float* pool) {
  int p = blockIdx.x * 256 + threadIdx.x;
  if (p >= L_IMG * HW_IMG) return;
  bool newmax = (ss[p] == mpss[p]) && !(pool[p] > 0.0f);
  mask[p] = ((mask[p] != 0.0f) || newmax) ? 1.0f : 0.0f;
}

__global__ __launch_bounds__(256) void k_final(const float* mask, const float* sc, float* o) {
  int p = blockIdx.x * 256 + threadIdx.x;
  if (p >= L_IMG * HW_IMG) return;
  o[p] = (mask[p] != 0.0f) ? sc[p] : 0.0f;
}

// ---------------------------------------------------------------------------
// K4: top-1024 indices per image via 256-bin key histogram (scores >= 0)
// ---------------------------------------------------------------------------
__global__ __launch_bounds__(256) void k_topk(const float* __restrict__ snms,
                                              int* __restrict__ topk) {
  __shared__ unsigned hist[256];
  __shared__ unsigned sB, sAbove, sSlots, sCntHi, sCntEq;
  int l = blockIdx.x, tid = threadIdx.x;
  const float* s = snms + l * HW_IMG;
  hist[tid] = 0;
  __syncthreads();
  for (int i = tid; i < HW_IMG; i += 256)
    atomicAdd(&hist[__float_as_uint(s[i]) >> 24], 1u);
  __syncthreads();
  if (tid == 0) {
    unsigned above = 0; int B = 0;
    for (int b = 255; b >= 0; --b) {
      if (above + hist[b] >= N_KPTS || b == 0) { B = b; break; }
      above += hist[b];
    }
    sB = (unsigned)B; sAbove = above; sSlots = N_KPTS - above;
    sCntHi = 0; sCntEq = 0;
  }
  __syncthreads();
  unsigned B = sB, above = sAbove, slots = sSlots;
  int* out = topk + l * N_KPTS;
  for (int i = tid; i < HW_IMG; i += 256) {
    unsigned bin = __float_as_uint(s[i]) >> 24;
    if (bin > B) {
      out[atomicAdd(&sCntHi, 1u)] = i;
    } else if (bin == B) {
      unsigned pos = atomicAdd(&sCntEq, 1u);
      if (pos < slots) out[above + pos] = i;
    }
  }
}

// ---------------------------------------------------------------------------
// K5: gather desc at keypoints + L2-normalize -> q[k][l][h]
// ---------------------------------------------------------------------------
__global__ __launch_bounds__(128) void k_gather(const float* __restrict__ desc,
                                                const int* __restrict__ topk,
                                                float* __restrict__ q) {
  __shared__ float red[128];
  int bid = blockIdx.x;
  int l = bid / N_KPTS, k = bid % N_KPTS;
  int h = threadIdx.x;
  int idx = topk[l * N_KPTS + k];
  float v = desc[(l * C_OUT + h) * HW_IMG + idx];
  red[h] = v * v;
  __syncthreads();
  for (int s = 64; s > 0; s >>= 1) {
    if (h < s) red[h] += red[h + s];
    __syncthreads();
  }
  float norm = fmaxf(sqrtf(red[0]), 1e-12f);
  q[(k * L_IMG + l) * C_OUT + h] = v / norm;
}

// ---------------------------------------------------------------------------
// K6: per-keypoint 6x6 attention + msg + proj -> pd[(l*3+o)][k]
// ---------------------------------------------------------------------------
__global__ __launch_bounds__(128) void k_attn(const float* __restrict__ q,
                                              const float* __restrict__ pw,
                                              const float* __restrict__ pb,
                                              float* __restrict__ pd) {
  __shared__ float qs[L_IMG][C_OUT];
  __shared__ float att[L_IMG][L_IMG];
  __shared__ float os[L_IMG][C_OUT];
  int k = blockIdx.x, h = threadIdx.x;
  for (int l = 0; l < L_IMG; ++l) qs[l][h] = q[(k * L_IMG + l) * C_OUT + h];
  __syncthreads();
  if (h < 36) {
    int a = h / 6, b = h % 6;
    float s = 0.0f;
    for (int c = 0; c < C_OUT; ++c) s += qs[a][c] * qs[b][c];
    att[a][b] = s * 0.08838834764831845f;  // 1/sqrt(128)
  }
  __syncthreads();
  if (h < 6) {
    float mx = -1e30f;
    for (int m = 0; m < 6; ++m) mx = fmaxf(mx, att[h][m]);
    float e[6], sum = 0.0f;
    for (int m = 0; m < 6; ++m) { e[m] = __expf(att[h][m] - mx); sum += e[m]; }
    for (int m = 0; m < 6; ++m) att[h][m] = e[m] / sum;
  }
  __syncthreads();
  for (int l = 0; l < L_IMG; ++l) {
    float m = 0.0f;
    for (int mm = 0; mm < 6; ++mm) m += att[l][mm] * qs[mm][h];
    os[l][h] = 2.0f * qs[l][h] + m;
  }
  __syncthreads();
  if (h < 18) {
    int l = h / 3, o = h % 3;
    float s = 0.0f;
    for (int c = 0; c < C_OUT; ++c) s += pw[o * C_OUT + c] * os[l][c];
    pd[(l * 3 + o) * N_KPTS + k] = s + pb[o];
  }
}

// ---------------------------------------------------------------------------
// K7: md[l][o] = min_k (pd[l][o][k] - pd[0][o][k])
// ---------------------------------------------------------------------------
__global__ __launch_bounds__(256) void k_minred(const float* __restrict__ pd,
                                                float* __restrict__ md) {
  __shared__ float red[256];
  int lo = blockIdx.x;
  int tid = threadIdx.x;
  int o = lo % 3;
  float m = 1e30f;
  for (int k = tid; k < N_KPTS; k += 256)
    m = fminf(m, pd[lo * N_KPTS + k] - pd[o * N_KPTS + k]);
  red[tid] = m;
  __syncthreads();
  for (int s = 128; s > 0; s >>= 1) {
    if (tid < s) red[tid] = fminf(red[tid], red[tid + s]);
    __syncthreads();
  }
  if (tid == 0) md[lo] = red[0];  // CORR_RATIO == 1.0
}

// ---------------------------------------------------------------------------
// K8: affine grid + bilinear grid_sample of feats -> out (L,256,H,W)
// ---------------------------------------------------------------------------
__global__ __launch_bounds__(256) void k_warp(const float* __restrict__ feats,
                                              const float* __restrict__ md,
                                              float* __restrict__ out) {
  int p = blockIdx.x * 256 + threadIdx.x;
  if (p >= L_IMG * HW_IMG) return;
  int l = p / HW_IMG, yx = p % HW_IMG;
  int y = yx / W_IMG, x = yx % W_IMG;
  float m0 = md[l * 3 + 0], m1 = md[l * 3 + 1], m2 = md[l * 3 + 2];
  float cth = __cosf(m2), sth = __sinf(m2);
  float gx = (x + 0.5f) * (2.0f / W_IMG) - 1.0f;
  float gy = (y + 0.5f) * (2.0f / H_IMG) - 1.0f;
  float u = cth * gx - sth * gy + m0;
  float v = sth * gx + cth * gy + m1;
  float ix = ((u + 1.0f) * W_IMG - 1.0f) * 0.5f;
  float iy = ((v + 1.0f) * H_IMG - 1.0f) * 0.5f;
  int ix0 = (int)floorf(ix), iy0 = (int)floorf(iy);
  int ix1 = ix0 + 1, iy1 = iy0 + 1;
  float wx1 = ix - (float)ix0, wx0 = 1.0f - wx1;
  float wy1 = iy - (float)iy0, wy0 = 1.0f - wy1;
  float v00 = (ix0 >= 0 && ix0 < W_IMG && iy0 >= 0 && iy0 < H_IMG) ? 1.0f : 0.0f;
  float v01 = (ix1 >= 0 && ix1 < W_IMG && iy0 >= 0 && iy0 < H_IMG) ? 1.0f : 0.0f;
  float v10 = (ix0 >= 0 && ix0 < W_IMG && iy1 >= 0 && iy1 < H_IMG) ? 1.0f : 0.0f;
  float v11 = (ix1 >= 0 && ix1 < W_IMG && iy1 >= 0 && iy1 < H_IMG) ? 1.0f : 0.0f;
  int cx0 = min(max(ix0, 0), W_IMG - 1), cx1 = min(max(ix1, 0), W_IMG - 1);
  int cy0 = min(max(iy0, 0), H_IMG - 1), cy1 = min(max(iy1, 0), H_IMG - 1);
  float w00 = wy0 * wx0 * v00, w01 = wy0 * wx1 * v01;
  float w10 = wy1 * wx0 * v10, w11 = wy1 * wx1 * v11;
  int o00 = cy0 * W_IMG + cx0, o01 = cy0 * W_IMG + cx1;
  int o10 = cy1 * W_IMG + cx0, o11 = cy1 * W_IMG + cx1;
  const float* base = feats + l * C_IN * HW_IMG;
  float* ob = out + l * C_IN * HW_IMG + yx;
  for (int c = 0; c < C_IN; ++c) {
    const float* fp = base + c * HW_IMG;
    ob[c * HW_IMG] = fp[o00] * w00 + fp[o01] * w01 + fp[o10] * w10 + fp[o11] * w11;
  }
}

// ---------------------------------------------------------------------------
extern "C" void kernel_launch(void* const* d_in, const int* in_sizes, int n_in,
                              void* d_out, int out_size, void* d_ws, size_t ws_size,
                              hipStream_t stream) {
  const float* feats = (const float*)d_in[0];
  const float* paw   = (const float*)d_in[1];
  const float* pab   = (const float*)d_in[2];
  const float* pbw   = (const float*)d_in[3];
  const float* pbb   = (const float*)d_in[4];
  const float* prw   = (const float*)d_in[5];
  const float* prb   = (const float*)d_in[6];

  char* ws = (char*)d_ws;
  size_t off = 0;
  auto bump = [&](size_t bytes) -> void* {
    void* p = ws + off;
    off += (bytes + 255) & ~(size_t)255;
    return p;
  };
  float*    desc   = (float*)bump((size_t)L_IMG * C_OUT * HW_IMG * 4);
  _Float16* wB     = (_Float16*)bump((size_t)9 * C_IN * C_OUT * 2);
  float*    scores = (float*)bump((size_t)L_IMG * HW_IMG * 4);
  float*    t1     = (float*)bump((size_t)L_IMG * HW_IMG * 4);
  float*    t2     = (float*)bump((size_t)L_IMG * HW_IMG * 4);
  float*    t3     = (float*)bump((size_t)L_IMG * HW_IMG * 4);
  float*    t4     = (float*)bump((size_t)L_IMG * HW_IMG * 4);
  float*    mask   = (float*)bump((size_t)L_IMG * HW_IMG * 4);
  float*    snms   = (float*)bump((size_t)L_IMG * HW_IMG * 4);
  int*      topk   = (int*)bump((size_t)L_IMG * N_KPTS * 4);
  float*    q      = (float*)bump((size_t)N_KPTS * L_IMG * C_OUT * 4);
  float*    pd     = (float*)bump((size_t)L_IMG * 3 * N_KPTS * 4);
  float*    md     = (float*)bump(256);

  int npx = L_IMG * HW_IMG;
  int nb  = (npx + 255) / 256;

  k_prep_w<<<(9 * C_IN * C_OUT + 255) / 256, 256, 0, stream>>>(paw, wB);
  k_conv_wmma<<<L_IMG * H_IMG * 4, 256, 0, stream>>>(feats, wB, pab, desc);
  k_scores<<<nb, 256, 0, stream>>>(desc, pbw, pbb, scores);

  // simple_nms(scores, 4) with 2 suppression iterations
  k_mp_h<<<nb, 256, 0, stream>>>(scores, t1);
  k_mp_v<<<nb, 256, 0, stream>>>(t1, t2);
  k_eq<<<nb, 256, 0, stream>>>(scores, t2, mask);
  for (int it = 0; it < 2; ++it) {
    k_mp_h<<<nb, 256, 0, stream>>>(mask, t1);
    k_mp_v<<<nb, 256, 0, stream>>>(t1, t2);          // supp pool
    k_supp<<<nb, 256, 0, stream>>>(scores, t2, t3);  // suppressed scores
    k_mp_h<<<nb, 256, 0, stream>>>(t3, t1);
    k_mp_v<<<nb, 256, 0, stream>>>(t1, t4);          // maxpool(ss)
    k_update<<<nb, 256, 0, stream>>>(mask, t3, t4, t2);
  }
  k_final<<<nb, 256, 0, stream>>>(mask, scores, snms);

  k_topk<<<L_IMG, 256, 0, stream>>>(snms, topk);
  k_gather<<<L_IMG * N_KPTS, 128, 0, stream>>>(desc, topk, q);
  k_attn<<<N_KPTS, 128, 0, stream>>>(q, prw, prb, pd);
  k_minred<<<L_IMG * 3, 256, 0, stream>>>(pd, md);
  k_warp<<<nb, 256, 0, stream>>>(feats, md, (float*)d_out);

  (void)in_sizes; (void)n_in; (void)out_size; (void)ws_size;
}
